// ModulatedDeformConv_4260607558277
// MI455X (gfx1250) — matrix operand order repaired and checked by reading, not compiled, not run on hardware
//
#include <hip/hip_runtime.h>
#include <hip/hip_bf16.h>

typedef _Float16 f16;
typedef __attribute__((ext_vector_type(16))) _Float16 v16h;
typedef __attribute__((ext_vector_type(8)))  _Float16 v8h;
typedef __attribute__((ext_vector_type(8)))  float    v8f;

#define BSZ  4
#define CIN  64
#define COUT 64
#define HD   128
#define WD   128
#define KT   9
#define NPIX (HD * WD)      // 16384
#define KDIM (CIN * KT)     // 576

// ---------------------------------------------------------------------------
// Pass 1: modulated-deformable im2col -> fp16 columns, layout [b][n][c*9+k]
// One thread per (b, output pixel). Bilinear corner indices and
// validity*mask*bilinear weights computed once per tap, reused over 64 ch.
// ---------------------------------------------------------------------------
__global__ __launch_bounds__(256) void dcn_im2col_kernel(
    const float* __restrict__ x, const float* __restrict__ off,
    const float* __restrict__ msk, f16* __restrict__ cols)
{
    int t  = blockIdx.x * blockDim.x + threadIdx.x;   // 0 .. B*NPIX-1
    int b  = t >> 14;
    int n  = t & (NPIX - 1);
    int ho = n >> 7;
    int wo = n & (WD - 1);

    int   idx[4][KT];
    float wgt[4][KT];

    const float* offb = off + (size_t)b * (2 * KT) * NPIX + n;
    const float* mskb = msk + (size_t)b * KT * NPIX + n;

#pragma unroll
    for (int k = 0; k < KT; ++k) {
        float dy = offb[(2 * k)     * NPIX];
        float dx = offb[(2 * k + 1) * NPIX];
        float mk = mskb[k * NPIX];
        float py = (float)(ho - 1 + k / 3) + dy;
        float px = (float)(wo - 1 + k % 3) + dx;
        float fy = floorf(py), fx = floorf(px);
        int   y0 = (int)fy,    x0 = (int)fx;
        float ly = py - fy,    lx = px - fx;
        int   y1 = y0 + 1,     x1 = x0 + 1;
        bool vy0 = (unsigned)y0 < (unsigned)HD;
        bool vy1 = (unsigned)y1 < (unsigned)HD;
        bool vx0 = (unsigned)x0 < (unsigned)WD;
        bool vx1 = (unsigned)x1 < (unsigned)WD;
        int cy0 = min(max(y0, 0), HD - 1), cy1 = min(max(y1, 0), HD - 1);
        int cx0 = min(max(x0, 0), WD - 1), cx1 = min(max(x1, 0), WD - 1);
        idx[0][k] = cy0 * WD + cx0;
        idx[1][k] = cy0 * WD + cx1;
        idx[2][k] = cy1 * WD + cx0;
        idx[3][k] = cy1 * WD + cx1;
        wgt[0][k] = (vy0 && vx0) ? (1.f - ly) * (1.f - lx) * mk : 0.f;
        wgt[1][k] = (vy0 && vx1) ? (1.f - ly) * lx         * mk : 0.f;
        wgt[2][k] = (vy1 && vx0) ? ly         * (1.f - lx) * mk : 0.f;
        wgt[3][k] = (vy1 && vx1) ? ly         * lx         * mk : 0.f;
    }

    const float* xb   = x + (size_t)b * CIN * NPIX;
    f16*         outp = cols + ((size_t)b * NPIX + n) * KDIM;

    for (int c = 0; c < CIN; ++c) {
        const float* xc = xb + c * NPIX;
        f16*         oc = outp + c * KT;
#pragma unroll
        for (int k = 0; k < KT; ++k) {
            float v = xc[idx[0][k]] * wgt[0][k]
                    + xc[idx[1][k]] * wgt[1][k]
                    + xc[idx[2][k]] * wgt[2][k]
                    + xc[idx[3][k]] * wgt[3][k];
            oc[k] = (f16)v;
        }
    }
}

// ---------------------------------------------------------------------------
// Weight fp32 -> fp16 (row-major [Cout][Cin*9], identical flat layout)
// ---------------------------------------------------------------------------
__global__ __launch_bounds__(256) void dcn_wcvt_kernel(
    const float* __restrict__ w, f16* __restrict__ wf)
{
    int i = blockIdx.x * blockDim.x + threadIdx.x;
    if (i < COUT * KDIM) wf[i] = (f16)w[i];
}

// ---------------------------------------------------------------------------
// Pass 2: GEMM via v_wmma_f32_16x16x32_f16.
// One wave computes the full 64(M) x 16(N) strip for one n-tile: the B
// fragment is loaded once per k-step and reused across 4 WMMAs (one per
// 16-row block of the weight matrix), quartering cols (L2) traffic.
//
// A fragment (16x32 f16): lane m = mt*16 + (lane&15); kg = lane>>4;
//   a[0..7]  = W[m][kb + kg*8 + i],  a[8..15] = W[m][kb + 16 + kg*8 + i].
// B fragment (32x16 f16): lane n = ntile*16 + (lane&15);
//   b[0..7]  = cols[b][n][kb + kg*8 + i],  b[8..15] = cols[b][n][kb + 16 + kg*8 + i].
// D (8 VGPRs): VGPR j holds M = mt*16 + 8*kg + j at column n.
// ---------------------------------------------------------------------------
__global__ __launch_bounds__(256) void dcn_gemm_kernel(
    const f16* __restrict__ wf, const f16* __restrict__ cols,
    const float* __restrict__ bias, float* __restrict__ out)
{
    int wid  = blockIdx.x * (blockDim.x >> 5) + (threadIdx.x >> 5);
    int lane = threadIdx.x & 31;
    int lo   = lane & 15;
    int kg   = lane >> 4;

    int ntile = wid & 1023;          // NPIX/16 = 1024
    int b     = wid >> 10;           // batch

    const f16* ap = wf + (size_t)lo * KDIM;                               // row lo
    const f16* bp = cols + ((size_t)b * NPIX + ntile * 16 + lo) * KDIM;   // col n

    v8f acc0 = {}, acc1 = {}, acc2 = {}, acc3 = {};

    for (int kb = 0; kb < KDIM; kb += 32) {
        v8h b0 = *(const v8h*)(bp + kb + kg * 8);
        v8h b1 = *(const v8h*)(bp + kb + kg * 8 + 16);
        v16h Bm = __builtin_shufflevector(b0, b1,
                    0,1,2,3,4,5,6,7,8,9,10,11,12,13,14,15);
        const f16* a = ap + kb + kg * 8;

        v8h a00 = *(const v8h*)(a);
        v8h a01 = *(const v8h*)(a + 16);
        v8h a10 = *(const v8h*)(a + 16 * KDIM);
        v8h a11 = *(const v8h*)(a + 16 * KDIM + 16);
        v8h a20 = *(const v8h*)(a + 32 * KDIM);
        v8h a21 = *(const v8h*)(a + 32 * KDIM + 16);
        v8h a30 = *(const v8h*)(a + 48 * KDIM);
        v8h a31 = *(const v8h*)(a + 48 * KDIM + 16);

        v16h A0 = __builtin_shufflevector(a00, a01,
                    0,1,2,3,4,5,6,7,8,9,10,11,12,13,14,15);
        v16h A1 = __builtin_shufflevector(a10, a11,
                    0,1,2,3,4,5,6,7,8,9,10,11,12,13,14,15);
        v16h A2 = __builtin_shufflevector(a20, a21,
                    0,1,2,3,4,5,6,7,8,9,10,11,12,13,14,15);
        v16h A3 = __builtin_shufflevector(a30, a31,
                    0,1,2,3,4,5,6,7,8,9,10,11,12,13,14,15);

        acc0 = __builtin_amdgcn_wmma_f32_16x16x32_f16(
                   false, A0, false, Bm, (short)0, acc0, false, false);
        acc1 = __builtin_amdgcn_wmma_f32_16x16x32_f16(
                   false, A1, false, Bm, (short)0, acc1, false, false);
        acc2 = __builtin_amdgcn_wmma_f32_16x16x32_f16(
                   false, A2, false, Bm, (short)0, acc2, false, false);
        acc3 = __builtin_amdgcn_wmma_f32_16x16x32_f16(
                   false, A3, false, Bm, (short)0, acc3, false, false);
    }

    // Epilogue: D layout -> out[b][m][n], fuse bias.
    float* ob = out + ((size_t)b * COUT + kg * 8) * NPIX + ntile * 16 + lo;
    const float* bq = bias + kg * 8;
    v8f accs[4] = {acc0, acc1, acc2, acc3};
#pragma unroll
    for (int mt = 0; mt < 4; ++mt) {
        float*       op = ob + (size_t)(mt * 16) * NPIX;
        const float* bm = bq + mt * 16;
#pragma unroll
        for (int j = 0; j < 8; ++j)
            op[(size_t)j * NPIX] = accs[mt][j] + bm[j];
    }
}

// ---------------------------------------------------------------------------
extern "C" void kernel_launch(void* const* d_in, const int* in_sizes, int n_in,
                              void* d_out, int out_size, void* d_ws, size_t ws_size,
                              hipStream_t stream) {
    (void)in_sizes; (void)n_in; (void)out_size; (void)ws_size;

    const float* x    = (const float*)d_in[0];
    const float* off  = (const float*)d_in[1];
    const float* msk  = (const float*)d_in[2];
    const float* w    = (const float*)d_in[3];
    const float* bias = (const float*)d_in[4];
    float*       out  = (float*)d_out;

    f16*   cols      = (f16*)d_ws;
    size_t colsBytes = (size_t)BSZ * NPIX * KDIM * sizeof(f16);  // 75,497,472 B
    f16*   wf        = (f16*)((char*)d_ws + colsBytes);          // 73,728 B

    dcn_im2col_kernel<<<(BSZ * NPIX) / 256, 256, 0, stream>>>(x, off, msk, cols);
    dcn_wcvt_kernel<<<(COUT * KDIM + 255) / 256, 256, 0, stream>>>(w, wf);
    // 4 batches * 1024 ntiles = 4096 waves / 8 waves per block = 512 blocks
    dcn_gemm_kernel<<<512, 256, 0, stream>>>(wf, cols, bias, out);
}
